// LGEB_89833535963786
// MI455X (gfx1250) — matrix operand order repaired and checked by reading, not compiled, not run on hardware
//
#include <hip/hip_runtime.h>
#include <hip/hip_bf16.h>

typedef _Float16 f16;
typedef __attribute__((ext_vector_type(2)))  _Float16 v2h;
typedef __attribute__((ext_vector_type(8)))  _Float16 v8h;
typedef __attribute__((ext_vector_type(16))) _Float16 v16h;
typedef __attribute__((ext_vector_type(8)))  float    v8f;

constexpr int NB = 64;    // batch
constexpr int NP = 128;   // particles
constexpr int NC = 72;    // channels
constexpr int KP = 96;    // K padded to 3x32
constexpr int NPAD = 80;  // N padded to 5x16
constexpr int K1P = 160;  // K padded for 144-wide layer (5x32)

// A-fragment: halves {0..7} at p[0], {16..23} at p[2] (16B-aligned)
__device__ inline v16h frag_a(const f16* p) {
  const v8h* q = (const v8h*)p;
  return __builtin_shufflevector(q[0], q[2],
      0, 1, 2, 3, 4, 5, 6, 7, 8, 9, 10, 11, 12, 13, 14, 15);
}
// B-fragment: halves {0..7} at p[0], {16..23} at p[2] (16B-aligned)
__device__ inline v16h frag_b(const f16* p) {
  const v8h* q = (const v8h*)p;
  return __builtin_shufflevector(q[0], q[2],
      0, 1, 2, 3, 4, 5, 6, 7, 8, 9, 10, 11, 12, 13, 14, 15);
}

// ---------------------------------------------------------------------------
// Kernel 1: A[b,i,c] = be1[c] + h[b,i,:]@We1[0:72,c];  Bv[b,j,c] = h@We1[72:144,c]
// ---------------------------------------------------------------------------
__global__ __launch_bounds__(256) void lgeb_pre(
    const float* __restrict__ h, const float* __restrict__ We1,
    const float* __restrict__ be1, float* __restrict__ A, float* __restrict__ Bv) {
  int idx = blockIdx.x * 256 + threadIdx.x;
  if (idx >= NB * NP * NC) return;
  int row = idx / NC, c = idx % NC;
  const float* hr = h + row * NC;
  float a = be1[c], bv = 0.f;
#pragma unroll 8
  for (int k = 0; k < NC; ++k) {
    float hv = hr[k];
    a  += hv * We1[k * NC + c];
    bv += hv * We1[(NC + k) * NC + c];
  }
  A[idx] = a;
  Bv[idx] = bv;
}

// ---------------------------------------------------------------------------
// Kernel 2: one workgroup per (b,i). Edge layer-2, gate dots, wm reduction,
//           x-update — all matrix work on WMMA.
// ---------------------------------------------------------------------------
__global__ __launch_bounds__(256) void lgeb_edge(
    const float* __restrict__ x,   const float* __restrict__ We1,
    const float* __restrict__ We2, const float* __restrict__ be2,
    const float* __restrict__ Wm,  const float* __restrict__ bm,
    const float* __restrict__ Wx,
    const float* A, const float* __restrict__ Bv,
    float* wmb, float* __restrict__ out_x) {

  __shared__ __align__(16) f16 lds_m1[NP * KP];       // m1 f16 (later m2 f16), A-layout
  __shared__ __align__(16) f16 lds_w2t[NPAD * KP];    // We2^T padded
  __shared__ __align__(16) f16 lds_m2t[NPAD * NP];    // m2^T [c][j] padded rows
  __shared__ __align__(16) f16 lds_gw[16 * KP];       // gate matrix^T: rows {Wm, Wx, 0..}
  __shared__ __align__(16) f16 lds_wjh[NP];           // w_j f16
  __shared__ float lds_A[NC], lds_wn[NC], lds_wp[NC], lds_be2[NC];
  __shared__ float4 lds_xj[NP];
  __shared__ float lds_pn[NP], lds_pp[NP], lds_phij[NP];

  const int tid = threadIdx.x;
  const int lane = tid & 31, wave = tid >> 5;
  const int b = blockIdx.x >> 7;
  const int i = blockIdx.x & 127;
  const int row_i = b * NP + i;
  const float bm0 = bm[0];

  // ---- phase 0 ----
  for (int t = tid; t < NC; t += 256) {
    lds_A[t]   = A[row_i * NC + t];
    lds_wn[t]  = We1[144 * NC + t];
    lds_wp[t]  = We1[145 * NC + t];
    lds_be2[t] = be2[t];
  }
  if (tid < NP) lds_xj[tid] = ((const float4*)x)[b * NP + tid];
  __syncthreads();

  // ---- phase 1: psi features; stage We2^T, gate matrix; zero m2t pad rows ----
  if (tid < NP) {
    float4 xi = lds_xj[i];
    float4 xj = lds_xj[tid];
    float d0 = xi.x - xj.x, d1 = xi.y - xj.y, d2 = xi.z - xj.z, d3 = xi.w - xj.w;
    float nr = d0 * d0 - d1 * d1 - d2 * d2 - d3 * d3;
    float pr = xi.x * xj.x - xi.y * xj.y - xi.z * xj.z - xi.w * xj.w;
    lds_pn[tid] = copysignf(log1pf(fabsf(nr)), nr);
    lds_pp[tid] = copysignf(log1pf(fabsf(pr)), pr);
  }
  for (int idx = tid; idx < NPAD * KP; idx += 256) {
    int n = idx / KP, k = idx % KP;
    lds_w2t[idx] = (f16)((n < NC && k < NC) ? We2[k * NC + n] : 0.f);
  }
  for (int idx = tid; idx < 16 * KP; idx += 256) {
    int n = idx / KP, k = idx % KP;
    float v = 0.f;
    if (k < NC) { if (n == 0) v = Wm[k]; else if (n == 1) v = Wx[k]; }
    lds_gw[idx] = (f16)v;
  }
  for (int t = tid; t < 8 * NP; t += 256)
    lds_m2t[(NC + t / NP) * NP + (t % NP)] = (f16)0.f;  // pad rows 72..79
  __syncthreads();

  // ---- phase 2: m1[j,c] = relu(A_i + B_j + pn*wn + pp*wp) ----
  for (int idx = tid; idx < NP * KP; idx += 256) {
    int j = idx / KP, c = idx % KP;
    float v = 0.f;
    if (c < NC) {
      v = fmaxf(lds_A[c] + Bv[(b * NP + j) * NC + c] +
                lds_pn[j] * lds_wn[c] + lds_pp[j] * lds_wp[c], 0.f);
    }
    lds_m1[idx] = (f16)v;
  }
  __syncthreads();

  const int mrow = wave * 16 + (lane & 15);
  const int ka = (lane >= 16) ? 8 : 0;
  const int ncol = lane & 15;
  const int kb = (lane < 16) ? 0 : 8;
  const int jl = wave * 16 + ((lane >= 16) ? 8 : 0);

  // ---- phase 3: m2 = relu(m1 @ We2 + be2); store f16 to lds_m1 (A-layout)
  //      and lds_m2t (B-layout) — each wave touches only its own rows ----
  {
    v16h af[3];
#pragma unroll
    for (int kk = 0; kk < 3; ++kk)
      af[kk] = frag_a(&lds_m1[mrow * KP + kk * 32 + ka]);
#pragma unroll
    for (int nt = 0; nt < 5; ++nt) {
      v8f acc = {};
#pragma unroll
      for (int kk = 0; kk < 3; ++kk) {
        v16h bf = frag_b(&lds_w2t[(nt * 16 + ncol) * KP + kk * 32 + kb]);
        acc = __builtin_amdgcn_wmma_f32_16x16x32_f16(
            false, af[kk], false, bf, (short)0, acc, false, false);
      }
      int n = nt * 16 + ncol;
      if (n < NC) {
        float bias = lds_be2[n];
        f16 hv[8];
#pragma unroll
        for (int r = 0; r < 8; ++r) {
          float v = fmaxf(acc[r] + bias, 0.f);
          hv[r] = (f16)v;
          lds_m1[(jl + r) * KP + n] = hv[r];
        }
#pragma unroll
        for (int q = 0; q < 4; ++q)
          *(v2h*)&lds_m2t[n * NP + jl + 2 * q] = (v2h){hv[2 * q], hv[2 * q + 1]};
      }
    }
  }

  // ---- phase 4: [sm|ph] = m2 @ [Wm|Wx] via WMMA (one n-tile) ----
  {
    v8f acc = {};
#pragma unroll
    for (int kk = 0; kk < 3; ++kk) {
      v16h af = frag_a(&lds_m1[mrow * KP + kk * 32 + ka]);
      v16h bf = frag_b(&lds_gw[ncol * KP + kk * 32 + kb]);
      acc = __builtin_amdgcn_wmma_f32_16x16x32_f16(
          false, af, false, bf, (short)0, acc, false, false);
    }
    if (ncol == 0) {        // N=0: sigmoid gate
#pragma unroll
      for (int r = 0; r < 8; ++r)
        lds_wjh[jl + r] = (f16)(1.f / (1.f + expf(-(acc[r] + bm0))));
    } else if (ncol == 1) { // N=1: phi_j
#pragma unroll
      for (int r = 0; r < 8; ++r)
        lds_phij[jl + r] = acc[r];
    }
  }
  __syncthreads();

  // ---- phase 5: wm = w(1x128) @ m2(128x72) via WMMA (waves 0..4);
  //      x-update on wave 5 ----
  if (wave < 5) {
    v8f acc = {};
#pragma unroll
    for (int kk = 0; kk < 4; ++kk) {
      v16h af = {};
      if ((lane & 15) == 0) {  // only M=0 row carries w
        int base = kk * 32 + ((lane >= 16) ? 8 : 0);
        af = frag_a(&lds_wjh[base]);
      }
      v16h bf = frag_b(&lds_m2t[(wave * 16 + ncol) * NP + kk * 32 + kb]);
      acc = __builtin_amdgcn_wmma_f32_16x16x32_f16(
          false, af, false, bf, (short)0, acc, false, false);
    }
    if (lane < 16) {           // D row M=0 -> wm
      int n = wave * 16 + lane;
      if (n < NC) wmb[row_i * NC + n] = acc[0];
    }
  } else if (wave == 5 && lane < 4) {
    int d = lane;
    float acc = 0.f;
    for (int j = 0; j < NP; ++j)
      acc += lds_phij[j] * ((const float*)&lds_xj[j])[d];
    out_x[row_i * 4 + d] = x[row_i * 4 + d] + 0.005f * (acc * (1.0f / NP));
  }
}

// ---------------------------------------------------------------------------
// Kernel 3: batched node MLP. h_out = h + relu([h|wm]@Wh1+bh1)@Wh2 + bh2
// 128 rows per block, full WMMA.
// ---------------------------------------------------------------------------
__global__ __launch_bounds__(256) void lgeb_node(
    const float* __restrict__ h,   const float* __restrict__ wmb,
    const float* __restrict__ Wh1, const float* __restrict__ bh1,
    const float* __restrict__ Wh2, const float* __restrict__ bh2,
    float* __restrict__ out_h) {

  // smem layout: [w1t 80*160 f16][w2t 80*96 f16][g 128*96 f16][cat 128*160 f16 | dh 128*73 f32]
  __shared__ __align__(16) char smem[65536 + 40960];
  f16*   w1t = (f16*)smem;                        // 25600 B
  f16*   w2t = (f16*)(smem + 25600);              // 15360 B
  f16*   g   = (f16*)(smem + 40960);              // 24576 B
  f16*   cat = (f16*)(smem + 65536);              // 40960 B
  float* dh  = (float*)(smem + 65536);            // 37376 B (aliases cat)
  __shared__ float lbh1[NPAD], lbh2[NPAD];

  const int tid = threadIdx.x;
  const int lane = tid & 31, wave = tid >> 5;
  const int r0 = blockIdx.x * NP;

  // ---- stage: weights (transposed, padded), biases, cat = [h|wm], g pad ----
  for (int idx = tid; idx < NPAD * K1P; idx += 256) {
    int n = idx / K1P, k = idx % K1P;
    w1t[idx] = (f16)((n < NC && k < 2 * NC) ? Wh1[k * NC + n] : 0.f);
  }
  for (int idx = tid; idx < NPAD * KP; idx += 256) {
    int n = idx / KP, k = idx % KP;
    w2t[idx] = (f16)((n < NC && k < NC) ? Wh2[k * NC + n] : 0.f);
  }
  for (int idx = tid; idx < NP * K1P; idx += 256) {
    int j = idx / K1P, k = idx % K1P;
    float v = 0.f;
    if (k < NC) v = h[(r0 + j) * NC + k];
    else if (k < 2 * NC) v = wmb[(r0 + j) * NC + (k - NC)];
    cat[idx] = (f16)v;
  }
  for (int idx = tid; idx < NP * KP; idx += 256) {
    int c = idx % KP;
    if (c >= NC) g[idx] = (f16)0.f;  // K-pad for layer 2
  }
  for (int t = tid; t < NPAD; t += 256) {
    lbh1[t] = (t < NC) ? bh1[t] : 0.f;
    lbh2[t] = (t < NC) ? bh2[t] : 0.f;
  }
  __syncthreads();

  const int mrow = wave * 16 + (lane & 15);
  const int ka = (lane >= 16) ? 8 : 0;
  const int ncol = lane & 15;
  const int kb = (lane < 16) ? 0 : 8;
  const int jl = wave * 16 + ((lane >= 16) ? 8 : 0);

  // ---- layer 1: g = relu(cat @ Wh1 + bh1), K=160 ----
  {
    v16h af[5];
#pragma unroll
    for (int kk = 0; kk < 5; ++kk)
      af[kk] = frag_a(&cat[mrow * K1P + kk * 32 + ka]);
#pragma unroll
    for (int nt = 0; nt < 5; ++nt) {
      v8f acc = {};
#pragma unroll
      for (int kk = 0; kk < 5; ++kk) {
        v16h bf = frag_b(&w1t[(nt * 16 + ncol) * K1P + kk * 32 + kb]);
        acc = __builtin_amdgcn_wmma_f32_16x16x32_f16(
            false, af[kk], false, bf, (short)0, acc, false, false);
      }
      int n = nt * 16 + ncol;
      if (n < NC) {
        float bias = lbh1[n];
#pragma unroll
        for (int r = 0; r < 8; ++r)
          g[(jl + r) * KP + n] = (f16)fmaxf(acc[r] + bias, 0.f);
      }
    }
  }
  __syncthreads();  // protects cat region before dh overwrites it

  // ---- layer 2: dh = g @ Wh2 + bh2, K=96 ----
  {
    v16h af[3];
#pragma unroll
    for (int kk = 0; kk < 3; ++kk)
      af[kk] = frag_a(&g[mrow * KP + kk * 32 + ka]);
#pragma unroll
    for (int nt = 0; nt < 5; ++nt) {
      v8f acc = {};
#pragma unroll
      for (int kk = 0; kk < 3; ++kk) {
        v16h bf = frag_b(&w2t[(nt * 16 + ncol) * KP + kk * 32 + kb]);
        acc = __builtin_amdgcn_wmma_f32_16x16x32_f16(
            false, af[kk], false, bf, (short)0, acc, false, false);
      }
      int n = nt * 16 + ncol;
      if (n < NC) {
        float bias = lbh2[n];
#pragma unroll
        for (int r = 0; r < 8; ++r)
          dh[(jl + r) * 73 + n] = acc[r] + bias;
      }
    }
  }
  __syncthreads();

  // ---- residual writeback (coalesced) ----
  for (int idx = tid; idx < NP * NC; idx += 256) {
    int j = idx / NC, c = idx % NC;
    out_h[(r0 + j) * NC + c] = h[(r0 + j) * NC + c] + dh[j * 73 + c];
  }
}

// ---------------------------------------------------------------------------
extern "C" void kernel_launch(void* const* d_in, const int* in_sizes, int n_in,
                              void* d_out, int out_size, void* d_ws, size_t ws_size,
                              hipStream_t stream) {
  const float* x   = (const float*)d_in[0];
  const float* h   = (const float*)d_in[1];
  const float* We1 = (const float*)d_in[2];
  const float* be1 = (const float*)d_in[3];
  const float* We2 = (const float*)d_in[4];
  const float* be2 = (const float*)d_in[5];
  const float* Wm  = (const float*)d_in[6];
  const float* bm  = (const float*)d_in[7];
  const float* Wh1 = (const float*)d_in[8];
  const float* bh1 = (const float*)d_in[9];
  const float* Wh2 = (const float*)d_in[10];
  const float* bh2 = (const float*)d_in[11];
  const float* Wx  = (const float*)d_in[12];

  size_t nelem = (size_t)NB * NP * NC;
  float* Aw  = (float*)d_ws;
  float* Bw  = Aw + nelem;
  // wm buffer: own region if workspace allows, else alias Aw (safe: A[row] is
  // consumed by row's own block before wm[row] is produced by that block).
  float* wmb = (ws_size >= 3 * nelem * sizeof(float)) ? (Bw + nelem) : Aw;

  float* out_h = (float*)d_out;                   // [NB*NP, 72]
  float* out_x = out_h + nelem;                   // [NB*NP, 4]

  int n1 = (int)nelem;
  lgeb_pre<<<(n1 + 255) / 256, 256, 0, stream>>>(h, We1, be1, Aw, Bw);
  lgeb_edge<<<NB * NP, 256, 0, stream>>>(x, We1, We2, be2, Wm, bm, Wx,
                                         Aw, Bw, wmb, out_x);
  lgeb_node<<<NB, 256, 0, stream>>>(h, wmb, Wh1, bh1, Wh2, bh2, out_h);
}
